// CooccurrenceEnhancer_21182778703904
// MI455X (gfx1250) — matrix operand order repaired and checked by reading, not compiled, not run on hardware
//
#include <hip/hip_runtime.h>

typedef __attribute__((ext_vector_type(16))) __bf16 v16bf_t;
typedef __attribute__((ext_vector_type(8)))  float  v8f_t;

#define E        512
#define CAND     32
#define ROWS     32          // batch rows per workgroup (2 WMMA M-tiles)
#define THREADS  256         // 8 wave32 waves
#define NEG_INF  (-__builtin_huge_valf())

// ---------------------------------------------------------------------------
// Kernel 1: transpose + fp32->bf16 convert cooc so B-fragments load as
// 16 contiguous bf16 (32B) per lane.  coocT[n*512 + k] = cooc[k*512 + n]
// ---------------------------------------------------------------------------
__global__ __launch_bounds__(256) void cooc_to_bf16_t(const float* __restrict__ cooc,
                                                      __bf16* __restrict__ coocT) {
  int idx = blockIdx.x * blockDim.x + threadIdx.x;     // 0 .. 512*512-1
  int k = idx >> 9;
  int n = idx & (E - 1);
  coocT[n * E + k] = (__bf16)cooc[k * E + n];
}

// ---------------------------------------------------------------------------
// Kernel 2: per 32-row tile:
//   scatter-add scores into sparse A tile (LDS, ds_add_f32 atomics)
//   WMMA bf16:  D[32x512] = A[32x512] * cooc[512x512]
//   mask existing ids with -inf, wave-cooperative top-32 of 512
// ---------------------------------------------------------------------------
__global__ __launch_bounds__(THREADS) void enhance_kernel(
    const int*    __restrict__ ids,        // [B,32] int32
    const float*  __restrict__ scores_in,  // [B,32]
    const __bf16* __restrict__ coocT,      // [512,512] (n-major)
    float*        __restrict__ out,        // ids: [0,B*64)  scores: [B*64,2*B*64)
    int B) {
  extern __shared__ float lds[];           // 32*512 fp32 = 64 KB (A tile, then scores)

  const int t       = threadIdx.x;
  const int lane    = t & 31;
  const int w       = t >> 5;              // wave 0..7
  const int rowBase = blockIdx.x * ROWS;
  const size_t scoreOut = (size_t)B * 64;  // offset of scores region in d_out

  // ---- zero A tile ----
  for (int i = t; i < ROWS * E; i += THREADS) lds[i] = 0.0f;
  __syncthreads();

  // ---- scatter-add candidates into sparse A tile + passthrough outputs ----
  for (int i = t; i < ROWS * CAND; i += THREADS) {
    int r  = i >> 5;
    int c  = i & 31;
    int gr = rowBase + r;
    int id = ids[gr * CAND + c];
    float s = scores_in[gr * CAND + c];
    atomicAdd(&lds[r * E + id], s);                        // ds_add_f32 (dup ids accumulate)
    out[(size_t)gr * 64 + c]            = (float)id;       // expanded_ids[:, :32]
    out[scoreOut + (size_t)gr * 64 + c] = s;               // expanded_scores[:, :32]
  }
  __syncthreads();

  // ---- WMMA matmul: each wave owns N-tiles [w*4 .. w*4+3], both M-tiles ----
  const int half = (lane < 16) ? 0 : 1;    // lane half selects K sub-block
  const int mrow = lane & 15;              // M (for A) / N (for B) within tile

  v8f_t acc[2][4] = {};                    // 2 M-tiles x 4 N-tiles, f32 accumulators

  for (int kc = 0; kc < E / 32; ++kc) {    // 16 K-chunks of 32
    // A fragments (contiguous k order, same as B): K(p) = half*16 + p
    v16bf_t afrag[2];
#pragma unroll
    for (int mt = 0; mt < 2; ++mt) {
      const float* ap = &lds[(mt * 16 + mrow) * E + kc * 32];
#pragma unroll
      for (int g = 0; g < 2; ++g) {
        const float* p = ap + half * 16 + g * 8;
#pragma unroll
        for (int j = 0; j < 8; ++j) afrag[mt][g * 8 + j] = (__bf16)p[j];
      }
    }
#pragma unroll
    for (int nt = 0; nt < 4; ++nt) {
      // B fragment (32x16 bf16): lane n = mrow, lanes0-15 K 0..15, lanes16-31 K 16..31
      int n = (w * 4 + nt) * 16 + mrow;
      const __bf16* bp = coocT + (size_t)n * E + kc * 32 + half * 16;
      v16bf_t bfrag = *(const v16bf_t*)bp;                 // 32B contiguous, aligned

      acc[0][nt] = __builtin_amdgcn_wmma_f32_16x16x32_bf16(
          false, afrag[0], false, bfrag, (short)0, acc[0][nt], false, false);
      acc[1][nt] = __builtin_amdgcn_wmma_f32_16x16x32_bf16(
          false, afrag[1], false, bfrag, (short)0, acc[1][nt], false, false);
    }
  }

  __syncthreads();   // everyone done reading A tile

  // ---- spill D tiles into LDS (C/D layout: VGPR r -> M = r + 8*halfl) ----
#pragma unroll
  for (int mt = 0; mt < 2; ++mt)
#pragma unroll
    for (int nt = 0; nt < 4; ++nt) {
      int col = (w * 4 + nt) * 16 + mrow;
#pragma unroll
      for (int r = 0; r < 8; ++r) {
        int row = mt * 16 + r + half * 8;
        lds[row * E + col] = acc[mt][nt][r];
      }
    }
  __syncthreads();

  // ---- mask existing candidates with -inf ----
  for (int i = t; i < ROWS * CAND; i += THREADS) {
    int r  = i >> 5;
    int id = ids[(rowBase + r) * CAND + (i & 31)];
    lds[r * E + id] = NEG_INF;
  }
  __syncthreads();

  // ---- top-32 of 512 per row; wave w handles rows w*4..w*4+3 ----
  for (int rr = 0; rr < 4; ++rr) {
    int r = w * 4 + rr;
    float vals[16];
#pragma unroll
    for (int j = 0; j < 16; ++j) vals[j] = lds[r * E + lane + 32 * j];

    // per-lane best (ascending scan keeps lowest index on ties)
    float bestV = vals[0];
    int   bestI = lane;
#pragma unroll
    for (int j = 1; j < 16; ++j) {
      float v = vals[j];
      if (v > bestV) { bestV = v; bestI = lane + 32 * j; }
    }

    for (int sel = 0; sel < 32; ++sel) {
      float wv = bestV;
      int   wi = bestI;
#pragma unroll
      for (int m = 16; m >= 1; m >>= 1) {            // wave32 argmax, 5 hops
        float ov = __shfl_xor(wv, m, 32);
        int   oi = __shfl_xor(wi, m, 32);
        if (ov > wv || (ov == wv && oi < wi)) { wv = ov; wi = oi; }
      }
      if (lane == 0) {
        size_t o = (size_t)(rowBase + r) * 64 + 32 + sel;
        out[o]            = (float)wi;               // additional_ids
        out[scoreOut + o] = wv;                      // additional_scores
      }
      if (lane == (wi & 31)) {                       // owner invalidates + rescans
        vals[wi >> 5] = NEG_INF;
        bestV = NEG_INF; bestI = lane;
#pragma unroll
        for (int j = 0; j < 16; ++j) {
          float v = vals[j];
          int ix = lane + 32 * j;
          if (v > bestV || (v == bestV && ix < bestI)) { bestV = v; bestI = ix; }
        }
      }
    }
  }
}

// ---------------------------------------------------------------------------
extern "C" void kernel_launch(void* const* d_in, const int* in_sizes, int n_in,
                              void* d_out, int out_size, void* d_ws, size_t ws_size,
                              hipStream_t stream) {
  const int*   ids       = (const int*)d_in[0];    // [B,32]
  const float* scores_in = (const float*)d_in[1];  // [B,32]
  const float* cooc      = (const float*)d_in[2];  // [512,512]
  float*       out       = (float*)d_out;
  __bf16*      coocT     = (__bf16*)d_ws;          // 512*512*2 = 512 KB

  const int B = in_sizes[0] / CAND;                // 65536

  // 1) transpose + convert cooc to bf16 (L2-resident thereafter)
  cooc_to_bf16_t<<<(E * E) / 256, 256, 0, stream>>>(cooc, coocT);

  // 2) main fused kernel: 32 rows per block, 64 KB dynamic LDS
  enhance_kernel<<<B / ROWS, THREADS, ROWS * E * sizeof(float), stream>>>(
      ids, scores_in, coocT, out, B);
}
